// DeepFM_29858612642272
// MI455X (gfx1250) — matrix operand order repaired and checked by reading, not compiled, not run on hardware
//
#include <hip/hip_runtime.h>
#include <hip/hip_bf16.h>
#include <stdint.h>

#define B_   16384
#define F_   26
#define V_   100000
#define D_   64
#define ND_  13
#define H1_  1024
#define H2_  512
#define H3_  256
#define K0_  ((F_ + 1) * D_)   // 1728, 54*32

typedef __attribute__((ext_vector_type(16))) _Float16 v16h;
typedef __attribute__((ext_vector_type(8)))  float    v8f;

union Frag16 { uint32_t u[8]; v16h v; };

// ---------------------------------------------------------------------------
// fp32 [K,N] row-major  ->  f16 [N,K] column-major (so K-pairs are contiguous)
// ---------------------------------------------------------------------------
__global__ void convert_transpose_f16(const float* __restrict__ W,
                                      _Float16* __restrict__ Wt,
                                      int K, int N) {
  long i = (long)blockIdx.x * blockDim.x + threadIdx.x;
  if (i >= (long)K * N) return;
  int n = (int)(i / K);
  int k = (int)(i % K);
  Wt[i] = (_Float16)W[(size_t)k * N + n];
}

// ---------------------------------------------------------------------------
// Per-row: dense@Wd, 26 embedding gathers -> h0 (f16), FM first+second order
// -> base[b] = bias + first + second   (one 64-thread block per batch row)
// ---------------------------------------------------------------------------
__global__ __launch_bounds__(64)
void prep_kernel(const float* __restrict__ dense,
                 const int*   __restrict__ sidx,
                 const float* __restrict__ bias,
                 const float* __restrict__ emb,
                 const float* __restrict__ lin,
                 const float* __restrict__ Wd,
                 const float* __restrict__ Wld,
                 const float* __restrict__ bld,
                 _Float16* __restrict__ h0,
                 float* __restrict__ base) {
  const int b = blockIdx.x;
  const int t = threadIdx.x;            // 0..63, one dim of D each
  __shared__ float red[64];
  __shared__ float dl[ND_];
  __shared__ int   idx[F_];
  if (t < ND_) dl[t]  = dense[(size_t)b * ND_ + t];
  if (t < F_)  idx[t] = sidx[(size_t)b * F_ + t];
  __syncthreads();

  // dense embedding: [13] @ [13,64]
  float de = 0.0f;
#pragma unroll
  for (int i = 0; i < ND_; ++i) de += dl[i] * Wd[i * D_ + t];

  _Float16* hrow = h0 + (size_t)b * K0_;
  hrow[t] = (_Float16)de;

  float sum_e = de, sumsq = de * de;
  for (int f = 0; f < F_; ++f) {
    float e = emb[((size_t)f * V_ + idx[f]) * D_ + t];
    hrow[(f + 1) * D_ + t] = (_Float16)e;
    sum_e += e;
    sumsq += e * e;
  }

  // second order partial (per dim) + first order partials (lane-parallel)
  float part = 0.5f * (sum_e * sum_e - sumsq);
  if (t < ND_)                  part += dl[t] * Wld[t];
  if (t >= ND_ && t < ND_ + F_) part += lin[(size_t)(t - ND_) * V_ + idx[t - ND_]];
  if (t == 0)                   part += bias[0] + bld[0];

  red[t] = part;
  __syncthreads();
  for (int s = 32; s > 0; s >>= 1) {
    if (t < s) red[t] += red[t + s];
    __syncthreads();
  }
  if (t == 0) base[b] = red[0];
}

// ---------------------------------------------------------------------------
// C[M,N] = relu(A[M,K] @ W[K,N]) in f16 with f32 WMMA accumulation.
// Bt is W in column-major [N,K]. 4 waves/block; each wave owns a 32(M)x64(N)
// tile (8 accumulators). Register ping-pong double-buffers the A/B fragments
// so the next k-step's global loads overlap the current 8 WMMAs.
// ---------------------------------------------------------------------------
template <int K, int N, int RELU>
__global__ __launch_bounds__(128)
void wmma_gemm_f16(const _Float16* __restrict__ A,
                   const _Float16* __restrict__ Bt,
                   _Float16* __restrict__ C) {
  static_assert((K % 64) == 0, "K must be an even number of 32-steps");
  const int lane = threadIdx.x & 31;
  const int wave = threadIdx.x >> 5;
  const int half = lane >> 4;           // K-half selector per ISA layout
  const int l16  = lane & 15;
  const int n0 = blockIdx.x * 64;
  const int m0 = blockIdx.y * 128 + wave * 32;

  v8f acc[8] = {};                      // acc[mi*4 + j]

  const _Float16* arow0 = A  + (size_t)(m0 + l16) * K;
  const _Float16* arow1 = A  + (size_t)(m0 + 16 + l16) * K;
  const _Float16* bcol  = Bt + (size_t)(n0 + l16) * K;

  Frag16 a[2][2], b[2][4];

  auto loadA = [&](int buf, int k0) {
#pragma unroll
    for (int v = 0; v < 8; ++v) {
      const int kb = (v < 4) ? (2 * v + 8 * half) : (16 + 2 * (v - 4) + 8 * half);
      a[buf][0].u[v] = *reinterpret_cast<const uint32_t*>(arow0 + k0 + kb);
      a[buf][1].u[v] = *reinterpret_cast<const uint32_t*>(arow1 + k0 + kb);
    }
  };
  auto loadB = [&](int buf, int k0) {
#pragma unroll
    for (int j = 0; j < 4; ++j) {
      const _Float16* bc = bcol + (size_t)(16 * j) * K + k0;
#pragma unroll
      for (int v = 0; v < 8; ++v)
        b[buf][j].u[v] = *reinterpret_cast<const uint32_t*>(bc + ((v < 4) ? (2 * v + 8 * half) : (16 + 2 * (v - 4) + 8 * half)));
    }
  };
  auto compute = [&](int buf) {
#pragma unroll
    for (int mi = 0; mi < 2; ++mi)
#pragma unroll
      for (int j = 0; j < 4; ++j)
        acc[mi * 4 + j] = __builtin_amdgcn_wmma_f32_16x16x32_f16(
            false, a[buf][mi].v, false, b[buf][j].v, (short)0,
            acc[mi * 4 + j], false, false);
  };

  loadA(0, 0);
  loadB(0, 0);
#pragma unroll 1
  for (int k0 = 0; k0 < K; k0 += 64) {
    loadA(1, k0 + 32);                  // prefetch next step into buf 1
    loadB(1, k0 + 32);
    compute(0);                         // overlap with in-flight loads
    if (k0 + 64 < K) {
      loadA(0, k0 + 64);                // prefetch step after next into buf 0
      loadB(0, k0 + 64);
    }
    compute(1);
  }

  // C/D layout: VGPR r, lane -> row = 16*mi + r + 8*half, col = 16*j + l16
#pragma unroll
  for (int mi = 0; mi < 2; ++mi)
#pragma unroll
    for (int j = 0; j < 4; ++j)
#pragma unroll
      for (int r = 0; r < 8; ++r) {
        float x = acc[mi * 4 + j][r];
        if (RELU) x = fmaxf(x, 0.0f);
        const int row = m0 + 16 * mi + r + 8 * half;
        const int col = n0 + 16 * j + l16;
        C[(size_t)row * N + col] = (_Float16)x;
      }
}

// ---------------------------------------------------------------------------
// out[b] = base[b] + h3[b,:] @ Wout   (one wave32 per row)
// ---------------------------------------------------------------------------
__global__ __launch_bounds__(256)
void final_kernel(const _Float16* __restrict__ h3,
                  const float* __restrict__ Wout,
                  const float* __restrict__ base,
                  float* __restrict__ out) {
  const int wavesPerBlock = blockDim.x >> 5;
  const int b = blockIdx.x * wavesPerBlock + (threadIdx.x >> 5);
  const int lane = threadIdx.x & 31;
  if (b >= B_) return;
  const _Float16* row = h3 + (size_t)b * H3_;
  float s = 0.0f;
#pragma unroll
  for (int i = lane; i < H3_; i += 32) s += (float)row[i] * Wout[i];
  for (int off = 16; off > 0; off >>= 1) s += __shfl_down(s, off, 32);
  if (lane == 0) out[b] = base[b] + s;
}

// ---------------------------------------------------------------------------
extern "C" void kernel_launch(void* const* d_in, const int* in_sizes, int n_in,
                              void* d_out, int out_size, void* d_ws, size_t ws_size,
                              hipStream_t stream) {
  const float* dense = (const float*)d_in[0];
  const int*   sidx  = (const int*)  d_in[1];
  const float* bias  = (const float*)d_in[2];
  const float* emb   = (const float*)d_in[3];
  const float* lin   = (const float*)d_in[4];
  const float* Wd    = (const float*)d_in[5];
  const float* Wld   = (const float*)d_in[6];
  const float* bld   = (const float*)d_in[7];
  const float* W1    = (const float*)d_in[8];
  const float* W2    = (const float*)d_in[9];
  const float* W3    = (const float*)d_in[10];
  const float* Wout  = (const float*)d_in[11];
  float* out = (float*)d_out;

  char* ws = (char*)d_ws;
  size_t off = 0;
  auto alloc = [&](size_t bytes) -> void* {
    void* p = ws + off;
    off += (bytes + 255) & ~(size_t)255;
    return p;
  };
  _Float16* h0  = (_Float16*)alloc((size_t)B_  * K0_ * sizeof(_Float16));
  _Float16* W1t = (_Float16*)alloc((size_t)K0_ * H1_ * sizeof(_Float16));
  _Float16* h1  = (_Float16*)alloc((size_t)B_  * H1_ * sizeof(_Float16));
  _Float16* W2t = (_Float16*)alloc((size_t)H1_ * H2_ * sizeof(_Float16));
  _Float16* h2  = (_Float16*)alloc((size_t)B_  * H2_ * sizeof(_Float16));
  _Float16* W3t = (_Float16*)alloc((size_t)H2_ * H3_ * sizeof(_Float16));
  _Float16* h3  = (_Float16*)alloc((size_t)B_  * H3_ * sizeof(_Float16));
  float*    base = (float*)  alloc((size_t)B_  * sizeof(float));

  // Weight conversion + transpose (tiny)
  { long n = (long)K0_ * H1_; convert_transpose_f16<<<(n + 255) / 256, 256, 0, stream>>>(W1, W1t, K0_, H1_); }
  { long n = (long)H1_ * H2_; convert_transpose_f16<<<(n + 255) / 256, 256, 0, stream>>>(W2, W2t, H1_, H2_); }
  { long n = (long)H2_ * H3_; convert_transpose_f16<<<(n + 255) / 256, 256, 0, stream>>>(W3, W3t, H2_, H3_); }

  // Gather + FM terms + h0 build
  prep_kernel<<<B_, 64, 0, stream>>>(dense, sidx, bias, emb, lin, Wd, Wld, bld, h0, base);

  // Deep MLP on WMMA (software-pipelined 32x64 tiles per wave)
  wmma_gemm_f16<K0_, H1_, 1><<<dim3(H1_ / 64, B_ / 128), 128, 0, stream>>>(h0, W1t, h1);
  wmma_gemm_f16<H1_, H2_, 1><<<dim3(H2_ / 64, B_ / 128), 128, 0, stream>>>(h1, W2t, h2);
  wmma_gemm_f16<H2_, H3_, 1><<<dim3(H3_ / 64, B_ / 128), 128, 0, stream>>>(h2, W3t, h3);

  // Final dot + combine with FM terms
  final_kernel<<<B_ / 8, 256, 0, stream>>>(h3, Wout, base, out);
}
